// GATE_GRUConv_InterMol_55516747268875
// MI455X (gfx1250) — compile-verified, matrix operand order checked by reading, not run on hardware
//
#include <hip/hip_runtime.h>
#include <math.h>

typedef _Float16 f16;
typedef __attribute__((ext_vector_type(16))) _Float16 v16h;
typedef __attribute__((ext_vector_type(8)))  float    v8f;

#define NN 50000
#define NE 250000
constexpr float PI_F = 3.14159265358979f;
constexpr float CUT  = 10.0f;

// ---------------- shared-memory layout constants (host+device) -------------
constexpr size_t WMB = 128 * 128 * sizeof(f16);              // 32768 B one f16 weight mat

constexpr size_t NODE_FIX  = WMB + 2 * 128 * sizeof(float);  // weights + w0 + bias
constexpr size_t NODE_WV   = 4096 /*A f16*/ + 8192 /*S f32*/ + 1024 /*misc*/;
constexpr size_t NODE_SMEM = NODE_FIX + 8 * NODE_WV;         // 140288

constexpr size_t EDGE_FIX  = 3 * WMB + 5 * 128 * sizeof(float);
constexpr size_t EDGE_WV   = 4096 /*A*/ + 8192 /*S0*/ + 8192 /*S1*/ + 4096 /*S2 f16*/
                           + 1280 /*misc 320f*/ + 128 /*idx 32i*/;
constexpr size_t EDGE_SMEM = EDGE_FIX + 4 * EDGE_WV;         // 204800

constexpr size_t FIN_FIX   = 2 * WMB + 4 * 128 * sizeof(float);
constexpr size_t FIN_WV    = 4096 + 8192 + 1024;
constexpr size_t FIN_SMEM  = FIN_FIX + 8 * FIN_WV;           // 174080

struct GV { const float *Wv1,*bv1,*Wv2,*bv2,*Ws,*bs,*Wg,*bg; };

// ---------------- small device helpers -------------------------------------
__device__ __forceinline__ float leaky01(float x){ return x > 0.f ? x : 0.01f*x; }
__device__ __forceinline__ float sigm(float x){ return 1.f/(1.f+expf(-x)); }
__device__ __forceinline__ float eluf(float x){ return x > 0.f ? x : expf(x)-1.f; }

// VNLeakyReLU for a single vector channel with 1x1 direction weight w
__device__ __forceinline__ void vnlrelu1(float w, float x[3]) {
  float n2 = x[0]*x[0] + x[1]*x[1] + x[2]*x[2];
  float dot = w * n2;
  float den = w*w*n2 + 1e-6f;
  float fneg = 1.f - (dot * w) / den;          // x_neg = x * fneg
  float keep = (dot >= 0.f) ? 1.f : fneg;
  float s = 0.2f + 0.8f * keep;                // slope=0.2
  x[0] *= s; x[1] *= s; x[2] *= s;
}

// per-row dot of a 16x128 LDS strip with a 128-wide global weight row.
// lanes (2r,2r+1) both hold the row-r result.
__device__ __forceinline__ float row_dot16(const float* s, const float* w, int lane) {
  int r = lane >> 1, h = lane & 1;
  float acc = 0.f;
  for (int k = h*64; k < h*64 + 64; ++k) acc += s[r*128 + k] * w[k];
  acc += __shfl_xor(acc, 1, 32);
  return acc;
}

// --------- 16x128 strip GEMM: out(16x128,f32) = A(16x128,f16) * W^T(f16) ----
// wm layout: wm[n*128+k] = W[n][k] (n = out-feature, k = in-feature)
__device__ __forceinline__ void wmma_strip(const f16* __restrict__ a,
                                           const f16* __restrict__ wm,
                                           float* __restrict__ out, int lane) {
  const int m  = lane & 15;
  const int ks = (lane < 16) ? 0 : 8;   // K-half owned by this lane group
  const int rb = (lane < 16) ? 0 : 8;   // C/D row base for this lane group
  #pragma unroll
  for (int nt = 0; nt < 8; ++nt) {
    v8f acc = {};
    #pragma unroll
    for (int kt = 0; kt < 4; ++kt) {
      const int kb = kt*32 + ks;
      const f16* ap = a  + m*128 + kb;
      const f16* bp = wm + (nt*16 + m)*128 + kb;
      v16h A, B;
      #pragma unroll
      for (int j = 0; j < 8; ++j) {
        A[j] = ap[j];  A[j+8] = ap[16+j];
        B[j] = bp[j];  B[j+8] = bp[16+j];
      }
      acc = __builtin_amdgcn_wmma_f32_16x16x32_f16(false, A, false, B,
                                                   (short)0, acc, false, false);
    }
    #pragma unroll
    for (int i = 0; i < 8; ++i)
      out[(rb+i)*128 + nt*16 + m] = acc[i];
  }
}

// ---------------- node GVLinear kernel --------------------------------------
// MODE 0: gv_perceptron (leaky on s, VNLeakyReLU(wvn) on v)
// MODE 1: plain gv_linear, plus n2e = s_out @ Wn2e.T + bn2e per node
template<int MODE>
__global__ __launch_bounds__(256)
void gvnode_kernel(const float* __restrict__ xin, const float* __restrict__ vin,
                   GV gv, const float* __restrict__ wvn,
                   const float* __restrict__ Wn2e, const float* __restrict__ bn2e,
                   float* __restrict__ s_out, float* __restrict__ v_out,
                   float* __restrict__ n2e_out, int N) {
  extern __shared__ char smem[];
  f16*   wm = (f16*)smem;
  float* w0 = (float*)(smem + WMB);
  float* bb = w0 + 128;
  const int tid = threadIdx.x, lane = tid & 31, wave = tid >> 5;
  char* wb   = smem + NODE_FIX + (size_t)wave * NODE_WV;
  f16*   a    = (f16*)wb;
  float* sout = (float*)(wb + 4096);
  float* msc  = (float*)(wb + 4096 + 8192);   // [0..15]=vn [16..63]=vi [64..79]=gate

  for (int i = tid; i < 128*128; i += 256) {
    int n = i >> 7, k = i & 127;
    wm[i] = (f16)gv.Ws[n*129 + 1 + k];
  }
  for (int n = tid; n < 128; n += 256) { w0[n] = gv.Ws[n*129]; bb[n] = gv.bs[n]; }

  const int r0 = blockIdx.x * 128 + wave * 16;
  if (lane < 16) {
    int r = r0 + lane;
    float p0=0.f,p1=0.f,p2=0.f;
    if (r < N) { p0 = vin[r*3+0]; p1 = vin[r*3+1]; p2 = vin[r*3+2]; }
    float wv1 = gv.Wv1[0], b1 = gv.bv1[0];
    float v0 = wv1*p0+b1, v1 = wv1*p1+b1, v2 = wv1*p2+b1;
    msc[16+lane*3+0] = v0; msc[16+lane*3+1] = v1; msc[16+lane*3+2] = v2;
    msc[lane] = sqrtf(v0*v0+v1*v1+v2*v2 + 1e-12f);
  }
  for (int i = lane; i < 2048; i += 32) {
    int rr = i >> 7, k = i & 127, r = r0 + rr;
    a[i] = (f16)((r < N) ? xin[(size_t)r*128 + k] : 0.f);
  }
  __syncthreads();
  wmma_strip(a, wm, sout, lane);
  __syncthreads();
  for (int i = lane; i < 2048; i += 32) {
    int rr = i >> 7, k = i & 127;
    sout[i] += msc[rr]*w0[k] + bb[k];
  }
  __syncthreads();
  { float g = row_dot16(sout, gv.Wg, lane);
    if ((lane & 1) == 0) msc[64 + (lane>>1)] = sigm(g + gv.bg[0]); }
  if (MODE == 1) {
    float t = row_dot16(sout, Wn2e, lane);
    int r = r0 + (lane >> 1);
    if ((lane & 1) == 0 && r < N) n2e_out[r] = t + bn2e[0];
  }
  __syncthreads();
  if (lane < 16) {
    int r = r0 + lane;
    if (r < N) {
      float g = msc[64+lane], wv2 = gv.Wv2[0], b2 = gv.bv2[0];
      float x[3];
      for (int d = 0; d < 3; ++d) x[d] = g*(wv2*msc[16+lane*3+d] + b2);
      if (MODE == 0) vnlrelu1(wvn[0], x);
      v_out[r*3+0] = x[0]; v_out[r*3+1] = x[1]; v_out[r*3+2] = x[2];
    }
  }
  for (int i = lane; i < 2048; i += 32) {
    int rr = i >> 7, k = i & 127, r = r0 + rr;
    if (r < N) {
      float v = sout[i];
      if (MODE == 0) v = leaky01(v);
      s_out[(size_t)r*128 + k] = v;
    }
  }
}

// ---------------- fused edge kernel -----------------------------------------
struct EP {
  const float *pos_src,*pos_dst,*edge_attr;
  const int *ei;
  GV egv; const float *ewvn;
  const float *Wsca,*bsca,*We2n,*be2n,*Wev,*expw;
  GV ogv;
  const float *nsd,*nvd,*n2ed;   // msg.node applied to (xd,pd)
  const float *nss,*nvs,*n2es;   // msg.node applied to (xs,ps)
  float *agg_s,*agg_v;
  int E, N;
};

// misc float offsets (per-wave, 320 floats)
#define M_ED   0
#define M_VIE  16
#define M_VNE  64
#define M_EVA  80
#define M_E2N  128
#define M_CEN  144
#define M_VT   160
#define M_VNO  208
#define M_GATE 224
#define M_VIO  240

__global__ __launch_bounds__(128)
void edge_kernel(EP p) {
  extern __shared__ char smem[];
  f16* wmE = (f16*)smem;
  f16* wmS = (f16*)(smem + WMB);
  f16* wmO = (f16*)(smem + 2*WMB);
  float* w0E = (float*)(smem + 3*WMB);
  float* bbE = w0E + 128;
  float* bsc = w0E + 256;
  float* w0O = w0E + 384;
  float* bbO = w0E + 512;
  const int tid = threadIdx.x, lane = tid & 31, wave = tid >> 5;
  char* wb  = smem + EDGE_FIX + (size_t)wave * EDGE_WV;
  f16*   a   = (f16*)wb;
  float* S0  = (float*)(wb + 4096);
  float* S1  = (float*)(wb + 4096 + 8192);
  f16*   S2h = (f16*)(wb + 4096 + 16384);
  float* msc = (float*)(wb + 4096 + 16384 + 4096);
  int*  iidx = (int*)(wb + 4096 + 16384 + 4096 + 1280);

  for (int i = tid; i < 128*128; i += 128) {
    int n = i >> 7, k = i & 127;
    wmE[i] = (f16)p.egv.Ws[n*129 + 1 + k];
    wmS[i] = (f16)p.Wsca[n*128 + k];
    wmO[i] = (f16)p.ogv.Ws[n*129 + 1 + k];
  }
  for (int n = tid; n < 128; n += 128) {
    w0E[n] = p.egv.Ws[n*129]; bbE[n] = p.egv.bs[n];
    bsc[n] = p.bsca[n];
    w0O[n] = p.ogv.Ws[n*129]; bbO[n] = p.ogv.bs[n];
  }
  const int e0 = blockIdx.x * 64 + wave * 16;

  // 1) geometry, edge gv vector pre, envelope
  if (lane < 16) {
    int e = e0 + lane, r = 0, c = 0;
    float d0=0.f,d1=0.f,d2=0.f;
    if (e < p.E) {
      r = p.ei[e]; c = p.ei[p.E + e];
      d0 = p.pos_src[r*3+0] - p.pos_dst[c*3+0];
      d1 = p.pos_src[r*3+1] - p.pos_dst[c*3+1];
      d2 = p.pos_src[r*3+2] - p.pos_dst[c*3+2];
    }
    iidx[lane] = r; iidx[16+lane] = c;
    float n2 = d0*d0 + d1*d1 + d2*d2;
    float ed = sqrtf(n2 + 1e-12f);
    msc[M_ED+lane] = ed;
    float inv = 1.f / (sqrtf(n2) + 1e-7f);
    float ew  = p.expw[0];
    float u0 = d0*inv*ew, u1 = d1*inv*ew, u2 = d2*inv*ew;
    float wv1 = p.egv.Wv1[0], b1 = p.egv.bv1[0];
    float v0 = wv1*u0+b1, v1 = wv1*u1+b1, v2 = wv1*u2+b1;
    msc[M_VIE+lane*3+0] = v0; msc[M_VIE+lane*3+1] = v1; msc[M_VIE+lane*3+2] = v2;
    msc[M_VNE+lane] = sqrtf(v0*v0+v1*v1+v2*v2 + 1e-12f);
    float C = 0.5f * (cosf(ed * PI_F / CUT) + 1.f);
    if (ed > CUT) C = 0.f;
    msc[M_CEN+lane] = C;
  }
  __syncthreads();

  // 2) stage es_in = [gauss(64) | edge_attr(64)]
  const float STEP = CUT / 63.f;
  const float COEF = -0.5f / (STEP*STEP);
  for (int i = lane; i < 2048; i += 32) {
    int rr = i >> 7, k = i & 127, e = e0 + rr;
    float v = 0.f;
    if (e < p.E) {
      if (k < 64) { float dd = msc[M_ED+rr] - (float)k * STEP; v = expf(COEF*dd*dd); }
      else        { v = p.edge_attr[(size_t)e*64 + (k-64)]; }
    }
    a[i] = (f16)v;
  }
  __syncthreads();
  wmma_strip(a, wmE, S0, lane);          // edge gv GEMM
  __syncthreads();
  for (int i = lane; i < 2048; i += 32) {
    int rr = i >> 7, k = i & 127;
    S0[i] += msc[M_VNE+rr]*w0E[k] + bbE[k];
  }
  __syncthreads();
  { float g = row_dot16(S0, p.egv.Wg, lane);
    if ((lane & 1) == 0) msc[M_GATE + (lane>>1)] = sigm(g + p.egv.bg[0]); }
  __syncthreads();
  // 3) activate edge scalars, stage for Wsca; activate edge vector
  for (int i = lane; i < 2048; i += 32) {
    float v = leaky01(S0[i]); S0[i] = v; a[i] = (f16)v;
  }
  if (lane < 16) {
    float g = msc[M_GATE+lane], wv2 = p.egv.Wv2[0], b2 = p.egv.bv2[0];
    float x[3];
    for (int d = 0; d < 3; ++d) x[d] = g*(wv2*msc[M_VIE+lane*3+d] + b2);
    vnlrelu1(p.ewvn[0], x);
    for (int d = 0; d < 3; ++d) msc[M_EVA+lane*3+d] = x[d];
  }
  __syncthreads();
  { float t = row_dot16(S0, p.We2n, lane);
    if ((lane & 1) == 0) msc[M_E2N + (lane>>1)] = t + p.be2n[0]; }
  wmma_strip(a, wmS, S1, lane);          // esW = es @ Wsca.T
  __syncthreads();
  for (int i = lane; i < 2048; i += 32) S1[i] += bsc[i & 127];
  // 4) ms (dst side, idx = col): vector pre
  if (lane < 16) {
    int c = iidx[16+lane];
    float e2n = msc[M_E2N+lane], n2e = p.n2ed[c], wev = p.Wev[0];
    float wv1 = p.ogv.Wv1[0], b1 = p.ogv.bv1[0];
    float vo[3];
    for (int d = 0; d < 3; ++d) {
      float yv = e2n * p.nvd[c*3+d] + n2e * (wev * msc[M_EVA+lane*3+d]);
      vo[d] = wv1*yv + b1; msc[M_VIO+lane*3+d] = vo[d];
    }
    msc[M_VNO+lane] = sqrtf(vo[0]*vo[0]+vo[1]*vo[1]+vo[2]*vo[2] + 1e-12f);
  }
  __syncthreads();
  for (int i = lane; i < 2048; i += 32) {
    int rr = i >> 7, k = i & 127; int c = iidx[16+rr];
    a[i] = (f16)(p.nsd[(size_t)c*128 + k] * S1[i]);
  }
  __syncthreads();
  wmma_strip(a, wmO, S0, lane);          // out gv GEMM (ms)
  __syncthreads();
  for (int i = lane; i < 2048; i += 32) {
    int rr = i >> 7, k = i & 127;
    S0[i] += msc[M_VNO+rr]*w0O[k] + bbO[k];
  }
  __syncthreads();
  { float g = row_dot16(S0, p.ogv.Wg, lane);
    if ((lane & 1) == 0) msc[M_GATE + (lane>>1)] = sigm(g + p.ogv.bg[0]); }
  __syncthreads();
  for (int i = lane; i < 2048; i += 32) {
    int rr = i >> 7;
    S2h[i] = (f16)(S0[i] * msc[M_CEN+rr]);   // ms_s * C saved
  }
  if (lane < 16) {
    float g = msc[M_GATE+lane], wv2 = p.ogv.Wv2[0], b2 = p.ogv.bv2[0];
    float C = msc[M_CEN+lane];
    for (int d = 0; d < 3; ++d)
      msc[M_VT+lane*3+d] = g*(wv2*msc[M_VIO+lane*3+d] + b2) * C;   // ms_v * C
  }
  __syncthreads();
  // 5) md (src side, idx = row)
  if (lane < 16) {
    int r = iidx[lane];
    float e2n = msc[M_E2N+lane], n2e = p.n2es[r], wev = p.Wev[0];
    float wv1 = p.ogv.Wv1[0], b1 = p.ogv.bv1[0];
    float vo[3];
    for (int d = 0; d < 3; ++d) {
      float yv = e2n * p.nvs[r*3+d] + n2e * (wev * msc[M_EVA+lane*3+d]);
      vo[d] = wv1*yv + b1; msc[M_VIO+lane*3+d] = vo[d];
    }
    msc[M_VNO+lane] = sqrtf(vo[0]*vo[0]+vo[1]*vo[1]+vo[2]*vo[2] + 1e-12f);
  }
  __syncthreads();
  for (int i = lane; i < 2048; i += 32) {
    int rr = i >> 7, k = i & 127; int r = iidx[rr];
    a[i] = (f16)(p.nss[(size_t)r*128 + k] * S1[i]);
  }
  __syncthreads();
  wmma_strip(a, wmO, S0, lane);          // out gv GEMM (md)
  __syncthreads();
  for (int i = lane; i < 2048; i += 32) {
    int rr = i >> 7, k = i & 127;
    S0[i] += msc[M_VNO+rr]*w0O[k] + bbO[k];
  }
  __syncthreads();
  { float g = row_dot16(S0, p.ogv.Wg, lane);
    if ((lane & 1) == 0) msc[M_GATE + (lane>>1)] = sigm(g + p.ogv.bg[0]); }
  __syncthreads();
  // 6) combine, elu, segment-sum by row
  for (int i = lane; i < 2048; i += 32) {
    int rr = i >> 7, k = i & 127, e = e0 + rr;
    if (e < p.E) {
      int r = iidx[rr];
      float m = 0.5f * ((float)S2h[i] + S0[i]*msc[M_CEN+rr]);
      atomicAdd(&p.agg_s[(size_t)r*128 + k], eluf(m));
    }
  }
  if (lane < 16) {
    int e = e0 + lane;
    if (e < p.E) {
      int r = iidx[lane];
      float g = msc[M_GATE+lane], wv2 = p.ogv.Wv2[0], b2 = p.ogv.bv2[0];
      float C = msc[M_CEN+lane];
      for (int d = 0; d < 3; ++d) {
        float vmd = g*(wv2*msc[M_VIO+lane*3+d] + b2) * C;
        float m = 0.5f * (msc[M_VT+lane*3+d] + vmd);
        atomicAdd(&p.agg_v[r*3 + d], eluf(m));
      }
    }
  }
}

// ---------------- finalize kernel -------------------------------------------
struct FP {
  const float *xs,*ps,*agg_s,*agg_v;
  GV cgv, ogv;
  const float *lnsg,*lnsb,*lnvg,*lnvb,*actw;
  float *out; int N;
};

__global__ __launch_bounds__(256)
void fin_kernel(FP p) {
  extern __shared__ char smem[];
  f16* wmC = (f16*)smem;
  f16* wmO = (f16*)(smem + WMB);
  float* w0C = (float*)(smem + 2*WMB);
  float* bbC = w0C + 128;
  float* w0O = w0C + 256;
  float* bbO = w0C + 384;
  const int tid = threadIdx.x, lane = tid & 31, wave = tid >> 5;
  char* wb = smem + FIN_FIX + (size_t)wave * FIN_WV;
  f16*   a   = (f16*)wb;
  float* S0  = (float*)(wb + 4096);
  float* msc = (float*)(wb + 4096 + 8192);
  // misc: 0 vn, 16 vi(48), 64 gate(16), 80 mu(16), 96 rsig(16), 112 viO(48), 160 vnO(16)

  for (int i = tid; i < 128*128; i += 256) {
    int n = i >> 7, k = i & 127;
    wmC[i] = (f16)p.cgv.Ws[n*129 + 1 + k];
    wmO[i] = (f16)p.ogv.Ws[n*129 + 1 + k];
  }
  for (int n = tid; n < 128; n += 256) {
    w0C[n] = p.cgv.Ws[n*129]; bbC[n] = p.cgv.bs[n];
    w0O[n] = p.ogv.Ws[n*129]; bbO[n] = p.ogv.bs[n];
  }
  const int r0 = blockIdx.x * 128 + wave * 16;
  if (lane < 16) {
    int r = r0 + lane;
    float p0=0.f,p1=0.f,p2=0.f;
    if (r < p.N) { p0 = p.ps[r*3+0]; p1 = p.ps[r*3+1]; p2 = p.ps[r*3+2]; }
    float wv1 = p.cgv.Wv1[0], b1 = p.cgv.bv1[0];
    float v0 = wv1*p0+b1, v1 = wv1*p1+b1, v2 = wv1*p2+b1;
    msc[16+lane*3+0] = v0; msc[16+lane*3+1] = v1; msc[16+lane*3+2] = v2;
    msc[lane] = sqrtf(v0*v0+v1*v1+v2*v2 + 1e-12f);
  }
  for (int i = lane; i < 2048; i += 32) {
    int rr = i >> 7, k = i & 127, r = r0 + rr;
    a[i] = (f16)((r < p.N) ? p.xs[(size_t)r*128 + k] : 0.f);
  }
  __syncthreads();
  wmma_strip(a, wmC, S0, lane);          // centroid GEMM
  __syncthreads();
  for (int i = lane; i < 2048; i += 32) {
    int rr = i >> 7, k = i & 127;
    S0[i] += msc[rr]*w0C[k] + bbC[k];
  }
  __syncthreads();
  { float g = row_dot16(S0, p.cgv.Wg, lane);
    if ((lane & 1) == 0) msc[64 + (lane>>1)] = sigm(g + p.cgv.bg[0]); }
  __syncthreads();
  for (int i = lane; i < 2048; i += 32) {
    int rr = i >> 7, k = i & 127, r = r0 + rr;
    S0[i] += (r < p.N) ? p.agg_s[(size_t)r*128 + k] : 0.f;
  }
  __syncthreads();
  { int r = lane >> 1, h = lane & 1; float su = 0.f, sq = 0.f;
    for (int k = h*64; k < h*64 + 64; ++k) { float v = S0[r*128+k]; su += v; sq += v*v; }
    su += __shfl_xor(su, 1, 32); sq += __shfl_xor(sq, 1, 32);
    if (h == 0) { float mu = su/128.f; float var = sq/128.f - mu*mu;
      msc[80+r] = mu; msc[96+r] = rsqrtf(var + 1e-5f); } }
  __syncthreads();
  for (int i = lane; i < 2048; i += 32) {
    int rr = i >> 7, k = i & 127;
    float v = (S0[i] - msc[80+rr]) * msc[96+rr] * p.lnsg[k] + p.lnsb[k];
    v = leaky01(v);
    S0[i] = v; a[i] = (f16)v;
  }
  if (lane < 16) {
    int r = r0 + lane;
    float g = msc[64+lane], wv2 = p.cgv.Wv2[0], b2 = p.cgv.bv2[0];
    float x[3];
    for (int d = 0; d < 3; ++d) {
      float cv = g*(wv2*msc[16+lane*3+d] + b2);
      float av = (r < p.N) ? p.agg_v[r*3+d] : 0.f;
      x[d] = cv + av;
    }
    float mu = (x[0]+x[1]+x[2]) / 3.f;
    float var = ((x[0]-mu)*(x[0]-mu)+(x[1]-mu)*(x[1]-mu)+(x[2]-mu)*(x[2]-mu)) / 3.f;
    float rs = rsqrtf(var + 1e-5f);
    for (int d = 0; d < 3; ++d) x[d] = (x[d]-mu)*rs*p.lnvg[d] + p.lnvb[d];
    vnlrelu1(p.actw[0], x);
    float wv1 = p.ogv.Wv1[0], b1 = p.ogv.bv1[0];
    float vo[3];
    for (int d = 0; d < 3; ++d) { vo[d] = wv1*x[d]+b1; msc[112+lane*3+d] = vo[d]; }
    msc[160+lane] = sqrtf(vo[0]*vo[0]+vo[1]*vo[1]+vo[2]*vo[2] + 1e-12f);
  }
  __syncthreads();
  wmma_strip(a, wmO, S0, lane);          // out_transform GEMM
  __syncthreads();
  for (int i = lane; i < 2048; i += 32) {
    int rr = i >> 7, k = i & 127;
    S0[i] += msc[160+rr]*w0O[k] + bbO[k];
  }
  __syncthreads();
  { float g = row_dot16(S0, p.ogv.Wg, lane);
    if ((lane & 1) == 0) msc[64 + (lane>>1)] = sigm(g + p.ogv.bg[0]); }
  __syncthreads();
  for (int i = lane; i < 2048; i += 32) {
    int rr = i >> 7, k = i & 127, r = r0 + rr;
    if (r < p.N) p.out[(size_t)r*128 + k] = S0[i];
  }
  if (lane < 16) {
    int r = r0 + lane;
    if (r < p.N) {
      float g = msc[64+lane], wv2 = p.ogv.Wv2[0], b2 = p.ogv.bv2[0];
      for (int d = 0; d < 3; ++d)
        p.out[(size_t)p.N*128 + r*3 + d] = g*(wv2*msc[112+lane*3+d] + b2);
    }
  }
}

// ---------------- host launch -----------------------------------------------
extern "C" void kernel_launch(void* const* d_in, const int* in_sizes, int n_in,
                              void* d_out, int out_size, void* d_ws, size_t ws_size,
                              hipStream_t stream) {
  (void)in_sizes; (void)n_in; (void)out_size; (void)ws_size;
  auto F = [&](int i) { return (const float*)d_in[i]; };
  auto gvat = [&](int b) {
    GV g; g.Wv1=F(b); g.bv1=F(b+1); g.Wv2=F(b+2); g.bv2=F(b+3);
    g.Ws=F(b+4); g.bs=F(b+5); g.Wg=F(b+6); g.bg=F(b+7); return g;
  };
  // depth-first insertion-order flattening of setup_inputs()
  const float* x_src   = F(0); const float* pos_src = F(1);
  const float* x_dst   = F(2); const float* pos_dst = F(3);
  const float* edge_attr = F(4);
  GV gps = gvat(5);  const float* wvn_s = F(13);
  GV gpd = gvat(14); const float* wvn_d = F(22);
  GV gmn = gvat(23);                       // msg.node
  GV gme = gvat(31); const float* wvn_e = F(39);
  const float *Wsca=F(40),*bsca=F(41),*We2n=F(42),*be2n=F(43),
              *Wn2e=F(44),*bn2e=F(45),*Wev=F(46);
  GV gmo = gvat(47);                       // msg.out
  GV gce = gvat(55);                       // centroid
  GV got = gvat(63);                       // out_transform
  const float* expw = F(71); const float* actw = F(72);
  const float *lnsg=F(73),*lnsb=F(74),*lnvg=F(75),*lnvb=F(76);
  const int* ei = (const int*)d_in[77];

  const int N = NN, E = NE;
  float* ws = (float*)d_ws;
  size_t o = 0;
  float* xs = ws+o;      o += (size_t)N*128;
  float* ps = ws+o;      o += (size_t)N*3;
  float* xd = ws+o;      o += (size_t)N*128;
  float* pd = ws+o;      o += (size_t)N*3;
  float* msd_s = ws+o;   o += (size_t)N*128;
  float* msd_v = ws+o;   o += (size_t)N*3;
  float* msd_n = ws+o;   o += (size_t)N;
  float* mss_s = ws+o;   o += (size_t)N*128;
  float* mss_v = ws+o;   o += (size_t)N*3;
  float* mss_n = ws+o;   o += (size_t)N;
  float* agg_s = ws+o;   o += (size_t)N*128;
  float* agg_v = ws+o;   o += (size_t)N*3;

  hipMemsetAsync(agg_s, 0, (size_t)N*131*sizeof(float), stream);  // agg_s | agg_v

  dim3 nb((N + 127) / 128), eb((E + 63) / 64);
  gvnode_kernel<0><<<nb, 256, NODE_SMEM, stream>>>(x_src, pos_src, gps, wvn_s,
      nullptr, nullptr, xs, ps, nullptr, N);
  gvnode_kernel<0><<<nb, 256, NODE_SMEM, stream>>>(x_dst, pos_dst, gpd, wvn_d,
      nullptr, nullptr, xd, pd, nullptr, N);
  gvnode_kernel<1><<<nb, 256, NODE_SMEM, stream>>>(xd, pd, gmn, nullptr,
      Wn2e, bn2e, msd_s, msd_v, msd_n, N);
  gvnode_kernel<1><<<nb, 256, NODE_SMEM, stream>>>(xs, ps, gmn, nullptr,
      Wn2e, bn2e, mss_s, mss_v, mss_n, N);

  EP ep;
  ep.pos_src = pos_src; ep.pos_dst = pos_dst; ep.edge_attr = edge_attr; ep.ei = ei;
  ep.egv = gme; ep.ewvn = wvn_e;
  ep.Wsca = Wsca; ep.bsca = bsca; ep.We2n = We2n; ep.be2n = be2n;
  ep.Wev = Wev; ep.expw = expw;
  ep.ogv = gmo;
  ep.nsd = msd_s; ep.nvd = msd_v; ep.n2ed = msd_n;
  ep.nss = mss_s; ep.nvs = mss_v; ep.n2es = mss_n;
  ep.agg_s = agg_s; ep.agg_v = agg_v; ep.E = E; ep.N = N;
  edge_kernel<<<eb, 128, EDGE_SMEM, stream>>>(ep);

  FP fp;
  fp.xs = xs; fp.ps = ps; fp.agg_s = agg_s; fp.agg_v = agg_v;
  fp.cgv = gce; fp.ogv = got;
  fp.lnsg = lnsg; fp.lnsb = lnsb; fp.lnvg = lnvg; fp.lnvb = lnvb; fp.actw = actw;
  fp.out = (float*)d_out; fp.N = N;
  fin_kernel<<<nb, 256, FIN_SMEM, stream>>>(fp);
}
